// VNLayer_24927990186014
// MI455X (gfx1250) — compile-verified, matrix operand order checked by reading, not run on hardware
//
#include <hip/hip_runtime.h>
#include <hip/hip_bf16.h>
#include <math.h>

// ---------------------------------------------------------------------------
// Problem constants (from the reference)
// ---------------------------------------------------------------------------
#define N_NODES 20000
#define N_EDGES 240000
#define HID     256
#define SDIM    128
#define FEATD   480
#define CCH     4

typedef __attribute__((ext_vector_type(16))) __bf16 v16bf;
typedef __attribute__((ext_vector_type(8)))  float  v8f;

union Frag16 { v16bf v; uint4 q[2]; };

__device__ __forceinline__ float silu_f(float x) { return x / (1.0f + __expf(-x)); }

// ---------------------------------------------------------------------------
// WMMA helpers (bf16 in, f32 accumulate). Layouts per CDNA5 ISA:
//   A (16x32, 16-bit): lane m=lane&15; elems 0..7  = K[kb..kb+7],
//                      elems 8..15 = K[16+kb..16+kb+7], kb = 8*(lane>>4)
//   B mirrored with n = lane&15 reading rows of W^T (i.e. columns of W)
//   C/D (16x16 f32):   lane n=lane&15; acc[r] is element (M = r+8*(lane>>4), N=n)
// ---------------------------------------------------------------------------
__device__ __forceinline__ v16bf load_frag_row(const __bf16* base, int stride,
                                               int k0, int lane) {
  int m  = lane & 15;
  int kb = (lane >> 4) << 3;
  const __bf16* p = base + (size_t)m * stride + k0 + kb;
  Frag16 f;
  f.q[0] = *reinterpret_cast<const uint4*>(p);
  f.q[1] = *reinterpret_cast<const uint4*>(p + 16);
  return f.v;
}

__device__ __forceinline__ v16bf load_frag_col(const __bf16* wt, int kpad,
                                               int ntb, int k0, int lane) {
  int n  = lane & 15;
  int kb = (lane >> 4) << 3;
  const __bf16* p = wt + (size_t)(ntb + n) * kpad + k0 + kb;
  Frag16 f;
  f.q[0] = *reinterpret_cast<const uint4*>(p);
  f.q[1] = *reinterpret_cast<const uint4*>(p + 16);
  return f.v;
}

// One 16x16 output tile of  D = A(16xK) * W(Kx16) + bias   (K = 32*ksteps)
__device__ __forceinline__ v8f wmma_gemm_tile(const __bf16* A, int ka,
                                              const __bf16* Wt, int kpad,
                                              int ntb, int ksteps,
                                              float bias, int lane) {
  v8f acc;
#pragma unroll
  for (int i = 0; i < 8; ++i) acc[i] = bias;
  for (int s = 0; s < ksteps; ++s) {
    v16bf a = load_frag_row(A, ka, s * 32, lane);
    v16bf b = load_frag_col(Wt, kpad, ntb, s * 32, lane);
    acc = __builtin_amdgcn_wmma_f32_16x16x32_bf16(false, a, false, b,
                                                  (short)0, acc, false, false);
  }
  return acc;
}

__device__ __forceinline__ void store_tile_bf16(__bf16* out, int stride, int ntb,
                                                v8f acc, int lane, bool act) {
  int n = lane & 15, hi = lane >> 4;
#pragma unroll
  for (int r = 0; r < 8; ++r) {
    float v = acc[r];
    if (act) v = silu_f(v);
    out[(size_t)(r + 8 * hi) * stride + ntb + n] = (__bf16)v;
  }
}

// ---------------------------------------------------------------------------
// Workspace layout (bytes). Weight bf16 copies are transposed: [out_pad][K_pad]
// so B-fragments are contiguous 16B loads. All base offsets 128B aligned.
// ---------------------------------------------------------------------------
constexpr size_t OFF_NFS   = 0;                                       // N*128 bf16
constexpr size_t OFF_VNF   = OFF_NFS   + (size_t)N_NODES * 128 * 2;   // N*256 bf16
constexpr size_t OFF_WSC   = OFF_VNF   + (size_t)N_NODES * 256 * 2;   // 128x128
constexpr size_t OFF_WCM1  = OFF_WSC   + 128 * 128 * 2;               // 256x416
constexpr size_t OFF_WCM2  = OFF_WCM1  + 256 * 416 * 2;               // 256x256
constexpr size_t OFF_WVR1  = OFF_WCM2  + 256 * 256 * 2;
constexpr size_t OFF_WVR2  = OFF_WVR1  + 256 * 256 * 2;               // 16x256
constexpr size_t OFF_WRV1  = OFF_WVR2  + 16  * 256 * 2;
constexpr size_t OFF_WRV2  = OFF_WRV1  + 256 * 256 * 2;               // 16x256
constexpr size_t OFF_WFT1  = OFF_WRV2  + 16  * 256 * 2;
constexpr size_t OFF_WFT2  = OFF_WFT1  + 256 * 256 * 2;
constexpr size_t OFF_WF2T1 = OFF_WFT2  + 256 * 256 * 2;
constexpr size_t OFF_WF2T2 = OFF_WF2T1 + 256 * 256 * 2;               // 128x256
constexpr size_t OFF_WVF1  = OFF_WF2T2 + 128 * 256 * 2;
constexpr size_t OFF_WVF2  = OFF_WVF1  + 256 * 256 * 2;
constexpr size_t OFF_AGG   = OFF_WVF2  + 256 * 256 * 2;               // N*256 f32
constexpr size_t OFF_AVR   = OFF_AGG   + (size_t)N_NODES * 256 * 4;   // N*3 f32
constexpr size_t OFF_ARV   = OFF_AVR   + (size_t)N_NODES * 3   * 4;   // N*12 f32
constexpr size_t OFF_CNT   = OFF_ARV   + (size_t)N_NODES * 12  * 4;   // N f32

// ---------------------------------------------------------------------------
// Kernel 0: zero the (contiguous) accumulator block + bf16-convert vn_feat
// ---------------------------------------------------------------------------
__global__ void init_kernel(const float* __restrict__ vn_feat,
                            __bf16* __restrict__ vnf,
                            float* __restrict__ aggz) {
  size_t i0     = (size_t)blockIdx.x * blockDim.x + threadIdx.x;
  size_t stride = (size_t)gridDim.x * blockDim.x;
  const size_t nz = (size_t)N_NODES * 272;   // agg_msg + agg_vr + agg_rv + cnt
  for (size_t j = i0; j < nz; j += stride) aggz[j] = 0.0f;
  const size_t nc = (size_t)N_NODES * HID;
  for (size_t j = i0; j < nc; j += stride) vnf[j] = (__bf16)vn_feat[j];
}

// ---------------------------------------------------------------------------
// Kernel 1: transpose + convert one weight matrix f32[K][Dout] -> bf16[DoutPad][KPad]
// ---------------------------------------------------------------------------
__global__ void wconv_kernel(const float* __restrict__ src, __bf16* __restrict__ dst,
                             int K, int Dout, int KPad, int DoutPad) {
  int total  = KPad * DoutPad;
  int i      = blockIdx.x * blockDim.x + threadIdx.x;
  int stride = gridDim.x * blockDim.x;
  for (; i < total; i += stride) {
    int o = i / KPad, k = i % KPad;
    float v = (o < Dout && k < K) ? src[(size_t)k * Dout + o] : 0.0f;
    dst[i] = (__bf16)v;
  }
}

// ---------------------------------------------------------------------------
// Kernel 2: nfs = (node_feat[:, :128] @ w_scalar) / sqrt(128), stored bf16
// 128 threads = 4 waves, 16 rows per block, 8 col tiles (2 per wave), K=128
// ---------------------------------------------------------------------------
__global__ __launch_bounds__(128) void nfs_kernel(const float* __restrict__ node_feat,
                                                  const __bf16* __restrict__ wsc,
                                                  __bf16* __restrict__ nfs) {
  __shared__ __bf16 A[16 * 128];
  const int tid = threadIdx.x, lane = tid & 31, wid = tid >> 5;
  const int rb  = blockIdx.x * 16;
  {
    int nl = tid >> 3, sub = tid & 7;
    const float* src = node_feat + (size_t)(rb + nl) * FEATD;
#pragma unroll
    for (int i = 0; i < 16; ++i) {
      int col = sub * 16 + i;
      A[nl * 128 + col] = (__bf16)src[col];
    }
  }
  __syncthreads();
  const float s = 0.08838834764831845f;  // 1/sqrt(128)
  for (int t = wid; t < 8; t += 4) {
    v8f acc = wmma_gemm_tile(A, 128, wsc, 128, t * 16, 4, 0.0f, lane);
    int n = lane & 15, hi = lane >> 4;
#pragma unroll
    for (int r = 0; r < 8; ++r)
      nfs[(size_t)(rb + r + 8 * hi) * 128 + t * 16 + n] = (__bf16)(acc[r] * s);
  }
}

// ---------------------------------------------------------------------------
// Kernel 3: fused edge pipeline. 16 edges per block, 128 threads (4 waves).
//   gather -> geometry -> cm MLP -> vr/rv MLPs -> atomic scatter-add.
// ---------------------------------------------------------------------------
__global__ __launch_bounds__(128) void edge_kernel(
    const int* __restrict__ ei,
    const float* __restrict__ node_pos, const float* __restrict__ vn_pos,
    const __bf16* __restrict__ nfs, const __bf16* __restrict__ vnf,
    const __bf16* __restrict__ wcm1, const float* __restrict__ cm_b1,
    const __bf16* __restrict__ wcm2, const float* __restrict__ cm_b2,
    const __bf16* __restrict__ wvr1, const float* __restrict__ vr_b1,
    const __bf16* __restrict__ wvr2, const float* __restrict__ vr_b2,
    const __bf16* __restrict__ wrv1, const float* __restrict__ rv_b1,
    const __bf16* __restrict__ wrv2, const float* __restrict__ rv_b2,
    float* __restrict__ agg_msg, float* __restrict__ agg_vr,
    float* __restrict__ agg_rv, float* __restrict__ cnt) {
  __shared__ __bf16 X[16 * 416];    // concat[nfs|vnf|ip] (K padded 400->416)
  __shared__ __bf16 B1[16 * 256];
  __shared__ __bf16 B2[16 * 256];   // msg (bf16 copy)
  __shared__ float  dnl[16 * 12];
  __shared__ int    rowl[16];
  __shared__ float  wvl[16 * 16];   // w_vr (cols 0..3 valid)
  __shared__ float  wrl[16 * 16];   // w_rv

  const int tid = threadIdx.x, lane = tid & 31, wid = tid >> 5;
  const int eb  = blockIdx.x * 16;

  // ---- gather nfs[row][0..63] and vnf[col][0..127] into X ----
  {
    int el = tid >> 3, sub = tid & 7;
    int e = eb + el;
    int r = ei[e];
    int c = ei[N_EDGES + e];
    if (sub == 0) rowl[el] = r;
    const uint4* srow = reinterpret_cast<const uint4*>(nfs + (size_t)r * 128);
    reinterpret_cast<uint4*>(&X[el * 416])[sub] = srow[sub];
    const uint4* svn = reinterpret_cast<const uint4*>(vnf + (size_t)c * 256);
    uint4* dvn = reinterpret_cast<uint4*>(&X[el * 416 + 128]);
    dvn[sub]     = svn[sub];
    dvn[sub + 8] = svn[sub + 8];
  }
  // ---- fp32 per-edge geometry: d, normalized gram ip, dn ----
  if (tid < 16) {
    int e = eb + tid;
    int r = ei[e];
    int c = ei[N_EDGES + e];
    float p0 = node_pos[(size_t)r * 3 + 0];
    float p1 = node_pos[(size_t)r * 3 + 1];
    float p2 = node_pos[(size_t)r * 3 + 2];
    float d[CCH][3];
#pragma unroll
    for (int ch = 0; ch < CCH; ++ch) {
      d[ch][0] = p0 - vn_pos[(size_t)c * 12 + ch * 3 + 0];
      d[ch][1] = p1 - vn_pos[(size_t)c * 12 + ch * 3 + 1];
      d[ch][2] = p2 - vn_pos[(size_t)c * 12 + ch * 3 + 2];
    }
    float ip[16], nrm2 = 0.0f;
#pragma unroll
    for (int a = 0; a < CCH; ++a)
#pragma unroll
      for (int b = 0; b < CCH; ++b) {
        float s = d[a][0] * d[b][0] + d[a][1] * d[b][1] + d[a][2] * d[b][2];
        ip[a * 4 + b] = s;
        nrm2 += s * s;
      }
    float inrm = 1.0f / fmaxf(sqrtf(nrm2), 1e-8f);
#pragma unroll
    for (int i = 0; i < 16; ++i) X[tid * 416 + 384 + i] = (__bf16)(ip[i] * inrm);
#pragma unroll
    for (int i = 0; i < 16; ++i) X[tid * 416 + 400 + i] = (__bf16)0.0f;  // K pad
#pragma unroll
    for (int ch = 0; ch < CCH; ++ch) {
      float l  = sqrtf(d[ch][0] * d[ch][0] + d[ch][1] * d[ch][1] + d[ch][2] * d[ch][2]);
      float il = 1.0f / fmaxf(l, 1e-8f);
      dnl[tid * 12 + ch * 3 + 0] = d[ch][0] * il;
      dnl[tid * 12 + ch * 3 + 1] = d[ch][1] * il;
      dnl[tid * 12 + ch * 3 + 2] = d[ch][2] * il;
    }
  }
  __syncthreads();

  // ---- h1 = silu(X @ cm_w1 + b1) -> B1   (K = 416 -> 13 wmma steps) ----
  for (int t = wid; t < 16; t += 4) {
    int ntb = t * 16;
    v8f acc = wmma_gemm_tile(X, 416, wcm1, 416, ntb, 13, cm_b1[ntb + (lane & 15)], lane);
    store_tile_bf16(B1, 256, ntb, acc, lane, true);
  }
  __syncthreads();

  // ---- msg = h1 @ cm_w2 + b2 -> B2 (bf16) + f32 atomic scatter-add ----
  for (int t = wid; t < 16; t += 4) {
    int ntb = t * 16;
    v8f acc = wmma_gemm_tile(B1, 256, wcm2, 256, ntb, 8, cm_b2[ntb + (lane & 15)], lane);
    int n = lane & 15, hi = lane >> 4;
#pragma unroll
    for (int r8 = 0; r8 < 8; ++r8) {
      int m   = r8 + 8 * hi;
      float v = acc[r8];
      B2[m * 256 + ntb + n] = (__bf16)v;
      unsafeAtomicAdd(&agg_msg[(size_t)rowl[m] * 256 + ntb + n], v);
    }
  }
  __syncthreads();

  // ---- h_vr = silu(msg @ vr_w1 + b1) -> B1 ----
  for (int t = wid; t < 16; t += 4) {
    int ntb = t * 16;
    v8f acc = wmma_gemm_tile(B2, 256, wvr1, 256, ntb, 8, vr_b1[ntb + (lane & 15)], lane);
    store_tile_bf16(B1, 256, ntb, acc, lane, true);
  }
  __syncthreads();

  // ---- h_rv = silu(msg @ rv_w1 + b1) -> X (reused, stride 256);
  //      wave0 also finishes w_vr = h_vr @ vr_w2 + b2 (16x16, cols 0..3 valid)
  __bf16* HR = X;
  for (int t = wid; t < 16; t += 4) {
    int ntb = t * 16;
    v8f acc = wmma_gemm_tile(B2, 256, wrv1, 256, ntb, 8, rv_b1[ntb + (lane & 15)], lane);
    store_tile_bf16(HR, 256, ntb, acc, lane, true);
  }
  if (wid == 0) {
    int n = lane & 15;
    float bias = (n < 4) ? vr_b2[n] : 0.0f;
    v8f acc = wmma_gemm_tile(B1, 256, wvr2, 256, 0, 8, bias, lane);
    int hi = lane >> 4;
#pragma unroll
    for (int r8 = 0; r8 < 8; ++r8) wvl[(r8 + 8 * hi) * 16 + n] = acc[r8];
  }
  __syncthreads();

  // ---- w_rv = h_rv @ rv_w2 + b2 (single 16x16 tile) ----
  if (wid == 1) {
    int n = lane & 15;
    v8f acc = wmma_gemm_tile(HR, 256, wrv2, 256, 0, 8, rv_b2[n], lane);
    int hi = lane >> 4;
#pragma unroll
    for (int r8 = 0; r8 < 8; ++r8) wrl[(r8 + 8 * hi) * 16 + n] = acc[r8];
  }
  __syncthreads();

  // ---- FCTP contractions + scatter-add of vr_upd / rv_upd / cnt ----
  if (tid < 16) {
    const float inv = 0.5f;  // 1/sqrt(C)
    int e = tid;
    int r = rowl[e];
    float v0 = 0.f, v1 = 0.f, v2 = 0.f;
#pragma unroll
    for (int ch = 0; ch < CCH; ++ch) {
      float w = wvl[e * 16 + ch];
      v0 += w * dnl[e * 12 + ch * 3 + 0];
      v1 += w * dnl[e * 12 + ch * 3 + 1];
      v2 += w * dnl[e * 12 + ch * 3 + 2];
    }
    unsafeAtomicAdd(&agg_vr[(size_t)r * 3 + 0], inv * v0);
    unsafeAtomicAdd(&agg_vr[(size_t)r * 3 + 1], inv * v1);
    unsafeAtomicAdd(&agg_vr[(size_t)r * 3 + 2], inv * v2);
#pragma unroll
    for (int u = 0; u < CCH; ++u)
#pragma unroll
      for (int j = 0; j < 3; ++j) {
        float s = 0.0f;
#pragma unroll
        for (int ch = 0; ch < CCH; ++ch)
          s += wrl[e * 16 + ch * 4 + u] * dnl[e * 12 + ch * 3 + j];
        unsafeAtomicAdd(&agg_rv[(size_t)r * 12 + u * 3 + j], -inv * s);
      }
    unsafeAtomicAdd(&cnt[r], 1.0f);
  }
}

// ---------------------------------------------------------------------------
// Kernel 4: fused node update (ft -> f2t chain, vf branch). 16 nodes / block.
// ---------------------------------------------------------------------------
__global__ __launch_bounds__(128) void node_kernel(
    const float* __restrict__ node_feat, const float* __restrict__ vn_feat,
    const float* __restrict__ agg_msg, const float* __restrict__ cnt,
    const __bf16* __restrict__ wft1,  const float* __restrict__ ft_b1,
    const __bf16* __restrict__ wft2,  const float* __restrict__ ft_b2,
    const __bf16* __restrict__ wf2t1, const float* __restrict__ f2t_b1,
    const __bf16* __restrict__ wf2t2, const float* __restrict__ f2t_b2,
    const __bf16* __restrict__ wvf1,  const float* __restrict__ vf_b1,
    const __bf16* __restrict__ wvf2,  const float* __restrict__ vf_b2,
    float* __restrict__ out_nf, float* __restrict__ out_vnf) {
  __shared__ __bf16 X0[16 * 256];  // agg_feat
  __shared__ __bf16 Ba[16 * 256];
  __shared__ __bf16 Bb[16 * 256];
  const int tid = threadIdx.x, lane = tid & 31, wid = tid >> 5;
  const int nb  = blockIdx.x * 16;
  {
    int nl = tid >> 3, sub = tid & 7;
    int node = nb + nl;
    float ic = 1.0f / fmaxf(cnt[node], 1.0f);
    const float* src = agg_msg + (size_t)node * 256;
#pragma unroll
    for (int i = 0; i < 32; ++i) {
      int col = sub * 32 + i;
      X0[nl * 256 + col] = (__bf16)(src[col] * ic);
    }
  }
  __syncthreads();
  // H1 = silu(X0 @ ft_w1 + b1) -> Ba
  for (int t = wid; t < 16; t += 4)
    store_tile_bf16(Ba, 256, t * 16,
                    wmma_gemm_tile(X0, 256, wft1, 256, t * 16, 8,
                                   ft_b1[t * 16 + (lane & 15)], lane),
                    lane, true);
  __syncthreads();
  // X1 = H1 @ ft_w2 + b2 -> Bb
  for (int t = wid; t < 16; t += 4)
    store_tile_bf16(Bb, 256, t * 16,
                    wmma_gemm_tile(Ba, 256, wft2, 256, t * 16, 8,
                                   ft_b2[t * 16 + (lane & 15)], lane),
                    lane, false);
  __syncthreads();
  // H2 = silu(X1 @ f2t_w1 + b1) -> Ba
  for (int t = wid; t < 16; t += 4)
    store_tile_bf16(Ba, 256, t * 16,
                    wmma_gemm_tile(Bb, 256, wf2t1, 256, t * 16, 8,
                                   f2t_b1[t * 16 + (lane & 15)], lane),
                    lane, true);
  __syncthreads();
  // ds = H2 @ f2t_w2 + b2 (16x128) -> node_feat residual on scalar block
  for (int t = wid; t < 8; t += 4) {
    v8f acc = wmma_gemm_tile(Ba, 256, wf2t2, 256, t * 16, 8,
                             f2t_b2[t * 16 + (lane & 15)], lane);
    int n = lane & 15, hi = lane >> 4;
#pragma unroll
    for (int r8 = 0; r8 < 8; ++r8) {
      size_t node = nb + r8 + 8 * hi;
      int col = t * 16 + n;
      out_nf[node * FEATD + col] = node_feat[node * FEATD + col] + acc[r8];
    }
  }
  // H3 = silu(X0 @ vf_w1 + b1) -> Bb (same phase: reads Ba/X0, writes Bb)
  for (int t = wid; t < 16; t += 4)
    store_tile_bf16(Bb, 256, t * 16,
                    wmma_gemm_tile(X0, 256, wvf1, 256, t * 16, 8,
                                   vf_b1[t * 16 + (lane & 15)], lane),
                    lane, true);
  __syncthreads();
  // vn_feat_out = vn_feat + H3 @ vf_w2 + b2
  for (int t = wid; t < 16; t += 4) {
    v8f acc = wmma_gemm_tile(Bb, 256, wvf2, 256, t * 16, 8,
                             vf_b2[t * 16 + (lane & 15)], lane);
    int n = lane & 15, hi = lane >> 4;
#pragma unroll
    for (int r8 = 0; r8 < 8; ++r8) {
      size_t node = nb + r8 + 8 * hi;
      int col = t * 16 + n;
      out_vnf[node * HID + col] = vn_feat[node * HID + col] + acc[r8];
    }
  }
}

// ---------------------------------------------------------------------------
// Kernel 5: untouched feature columns + pos residuals
// ---------------------------------------------------------------------------
__global__ void tail_kernel(const float* __restrict__ node_feat,
                            const float* __restrict__ node_pos,
                            const float* __restrict__ vn_pos,
                            const float* __restrict__ agg_vr,
                            const float* __restrict__ agg_rv,
                            const float* __restrict__ cnt,
                            float* __restrict__ out_nf,
                            float* __restrict__ out_np,
                            float* __restrict__ out_vnp) {
  size_t i0     = (size_t)blockIdx.x * blockDim.x + threadIdx.x;
  size_t stride = (size_t)gridDim.x * blockDim.x;
  const size_t n1 = (size_t)N_NODES * (FEATD - SDIM);
  for (size_t j = i0; j < n1; j += stride) {
    size_t node = j / (FEATD - SDIM);
    int col = SDIM + (int)(j % (FEATD - SDIM));
    out_nf[node * FEATD + col] = node_feat[node * FEATD + col];
  }
  const size_t n2 = (size_t)N_NODES * 3;
  for (size_t j = i0; j < n2; j += stride) {
    float ic = 1.0f / fmaxf(cnt[j / 3], 1.0f);
    out_np[j] = node_pos[j] + agg_vr[j] * ic;
  }
  const size_t n3 = (size_t)N_NODES * 12;
  for (size_t j = i0; j < n3; j += stride) {
    float ic = 1.0f / fmaxf(cnt[j / 12], 1.0f);
    out_vnp[j] = vn_pos[j] + agg_rv[j] * ic;
  }
}

// ---------------------------------------------------------------------------
// Host launcher
// ---------------------------------------------------------------------------
extern "C" void kernel_launch(void* const* d_in, const int* in_sizes, int n_in,
                              void* d_out, int out_size, void* d_ws, size_t ws_size,
                              hipStream_t stream) {
  const float* node_feat = (const float*)d_in[0];
  const float* node_pos  = (const float*)d_in[1];
  const float* vn_feat   = (const float*)d_in[2];
  const float* vn_pos    = (const float*)d_in[3];
  const int*   edge_idx  = (const int*)d_in[4];

  char* ws = (char*)d_ws;
  __bf16* nfs    = (__bf16*)(ws + OFF_NFS);
  __bf16* vnf    = (__bf16*)(ws + OFF_VNF);
  float*  aggm   = (float*)(ws + OFF_AGG);
  float*  aggvr  = (float*)(ws + OFF_AVR);
  float*  aggrv  = (float*)(ws + OFF_ARV);
  float*  cntp   = (float*)(ws + OFF_CNT);

  float* out_nf  = (float*)d_out;
  float* out_np  = out_nf  + (size_t)N_NODES * FEATD;
  float* out_vnf = out_np  + (size_t)N_NODES * 3;
  float* out_vnp = out_vnf + (size_t)N_NODES * HID;

  // 0) zero accumulators + bf16 copy of vn_feat
  init_kernel<<<2048, 256, 0, stream>>>(vn_feat, vnf, aggm);

  // 1) transpose+convert every weight matrix to bf16 [out_pad][K_pad]
  auto wc = [&](int in_idx, size_t off, int K, int Dout, int KPad, int DoutPad) {
    int total  = KPad * DoutPad;
    int blocks = (total + 255) / 256;
    wconv_kernel<<<blocks, 256, 0, stream>>>((const float*)d_in[in_idx],
                                             (__bf16*)(ws + off),
                                             K, Dout, KPad, DoutPad);
  };
  wc(5,  OFF_WSC,   128, 128, 128, 128);   // w_scalar
  wc(6,  OFF_WCM1,  400, 256, 416, 256);   // cm_w1
  wc(8,  OFF_WCM2,  256, 256, 256, 256);   // cm_w2
  wc(10, OFF_WVR1,  256, 256, 256, 256);   // vr_w1
  wc(12, OFF_WVR2,  256,   4, 256,  16);   // vr_w2 (out padded to 16)
  wc(14, OFF_WRV1,  256, 256, 256, 256);   // rv_w1
  wc(16, OFF_WRV2,  256,  16, 256,  16);   // rv_w2
  wc(18, OFF_WFT1,  256, 256, 256, 256);   // ft_w1
  wc(20, OFF_WFT2,  256, 256, 256, 256);   // ft_w2
  wc(22, OFF_WF2T1, 256, 256, 256, 256);   // f2t_w1
  wc(24, OFF_WF2T2, 256, 128, 256, 128);   // f2t_w2
  wc(26, OFF_WVF1,  256, 256, 256, 256);   // vf_w1
  wc(28, OFF_WVF2,  256, 256, 256, 256);   // vf_w2

  // 2) nfs GEMM
  nfs_kernel<<<N_NODES / 16, 128, 0, stream>>>(node_feat, (__bf16*)(ws + OFF_WSC), nfs);

  // 3) fused edge pipeline
  edge_kernel<<<N_EDGES / 16, 128, 0, stream>>>(
      edge_idx, node_pos, vn_pos, nfs, vnf,
      (__bf16*)(ws + OFF_WCM1), (const float*)d_in[7],
      (__bf16*)(ws + OFF_WCM2), (const float*)d_in[9],
      (__bf16*)(ws + OFF_WVR1), (const float*)d_in[11],
      (__bf16*)(ws + OFF_WVR2), (const float*)d_in[13],
      (__bf16*)(ws + OFF_WRV1), (const float*)d_in[15],
      (__bf16*)(ws + OFF_WRV2), (const float*)d_in[17],
      aggm, aggvr, aggrv, cntp);

  // 4) fused node update
  node_kernel<<<N_NODES / 16, 128, 0, stream>>>(
      node_feat, vn_feat, aggm, cntp,
      (__bf16*)(ws + OFF_WFT1),  (const float*)d_in[19],
      (__bf16*)(ws + OFF_WFT2),  (const float*)d_in[21],
      (__bf16*)(ws + OFF_WF2T1), (const float*)d_in[23],
      (__bf16*)(ws + OFF_WF2T2), (const float*)d_in[25],
      (__bf16*)(ws + OFF_WVF1),  (const float*)d_in[27],
      (__bf16*)(ws + OFF_WVF2),  (const float*)d_in[29],
      out_nf, out_vnf);

  // 5) residual/copy tail
  tail_kernel<<<2048, 256, 0, stream>>>(node_feat, node_pos, vn_pos,
                                        aggvr, aggrv, cntp,
                                        out_nf, out_np, out_vnp);
}
